// JointLSTM_1262720385110
// MI455X (gfx1250) — matrix operand order repaired and checked by reading, not compiled, not run on hardware
//
#include <hip/hip_runtime.h>
#include <hip/hip_bf16.h>

// ---------------- model dimensions ----------------
constexpr int B_   = 128;
constexpr int A_   = 30;
constexpr int T_   = 65;
constexpr int FIN_ = 4;
constexpr int HID_ = 64;
constexpr int H_   = 1920;      // A_*HID_
constexpr int NSTEP = T_ - 1;   // 64
constexpr int KT   = H_ / 32;   // 60 k-tiles of 32
constexpr int MT   = B_ / 16;   // 8 m-tiles
constexpr int NT_H = H_ / 16;   // 120 n-tiles per gate / for W3
constexpr int NT_G = 4 * NT_H;  // 480 n-tiles for Wih/Whh (4 gates)

typedef __bf16 bf16_t;
typedef __attribute__((ext_vector_type(16))) __bf16 v16bf;
typedef __attribute__((ext_vector_type(8)))  __bf16 v8bf;
typedef __attribute__((ext_vector_type(8)))  float  v8f;

// ---------------- helpers ----------------
__device__ __forceinline__ unsigned short f2bf(float f) {
  unsigned u = __builtin_bit_cast(unsigned, f);
  u += 0x7fffu + ((u >> 16) & 1u);           // round-to-nearest-even
  return (unsigned short)(u >> 16);
}
__device__ __forceinline__ float bf2f(unsigned short h) {
  unsigned u = ((unsigned)h) << 16;
  return __builtin_bit_cast(float, u);
}
__device__ __forceinline__ float sigmoidf_(float x) {
  return 1.0f / (1.0f + __expf(-x));
}

// Load a 16x32 bf16 A-fragment from a row-major [rows][H_] bf16 matrix.
// A layout used here: lanes 0-15 / 16-31 both hold rows m=lane&15; halves 0..15 are
// K = kt*32 + hi*16 + e (contiguous, matching swizzle_w B fragments).
__device__ __forceinline__ v16bf load_a(const bf16_t* __restrict__ base,
                                        int row, int kt, int hi) {
  const bf16_t* p = base + (size_t)row * H_ + kt * 32 + hi * 16;
  v8bf lo = *(const v8bf*)p;
  v8bf hh = *(const v8bf*)(p + 8);
  v16bf a;
#pragma unroll
  for (int i = 0; i < 8; ++i) { a[i] = lo[i]; a[i + 8] = hh[i]; }
  return a;
}

// ---------------- one-time kernels ----------------
// Convert fp32 weight W[nrows][H_] into bf16 WMMA B-fragments.
// Fragment (nt,kt): frag[lane][e] = W[nt*16 + (lane&15)][kt*32 + (lane>>4)*16 + e]
// stored contiguously: dst[((nt*KT + kt)*512) + lane*16 + e]
__global__ void swizzle_w(const float* __restrict__ src,
                          bf16_t* __restrict__ dst, int ntiles) {
  int tid  = blockIdx.x * blockDim.x + threadIdx.x;
  int frag = tid >> 5, lane = tid & 31;
  int nt = frag / KT, kt = frag % KT;
  if (nt >= ntiles) return;
  int row = nt * 16 + (lane & 15);
  int k0  = kt * 32 + (lane >> 4) * 16;
  const float* s = src + (size_t)row * H_ + k0;
  unsigned short* d = (unsigned short*)dst + (size_t)frag * 512 + lane * 16;
#pragma unroll
  for (int e = 0; e < 16; ++e) d[e] = f2bf(s[e]);
}

__global__ void zero_ws(unsigned int* __restrict__ p, size_t n) {
  for (size_t i = (size_t)blockIdx.x * blockDim.x + threadIdx.x; i < n;
       i += (size_t)gridDim.x * blockDim.x)
    p[i] = 0u;
}

// ---------------- per-step kernels ----------------
// Per-atom MLP with burn-in / feedback selection. One block per (b,a).
__global__ void __launch_bounds__(HID_) mlp_kernel(
    const float* __restrict__ inputs, const float* __restrict__ prev,
    const float* __restrict__ W1, const float* __restrict__ b1,
    const float* __restrict__ W2, const float* __restrict__ b2,
    bf16_t* __restrict__ xbuf, float* __restrict__ insbuf,
    const int* __restrict__ burn, int t) {
  __shared__ float s_ins[FIN_];
  __shared__ float s_x1[HID_];
  int ba = blockIdx.x;          // b*A + a
  int j  = threadIdx.x;         // 0..63
  int bq = ba / A_, aq = ba % A_;
  if (j < FIN_) {
    float gt = inputs[(((size_t)bq * A_ + aq) * T_ + t) * FIN_ + j];
    float pv = prev[(size_t)ba * FIN_ + j];
    float v  = (t <= *burn) ? gt : pv;
    s_ins[j] = v;
    insbuf[(size_t)ba * FIN_ + j] = v;
  }
  __syncthreads();
  float acc = b1[j];
#pragma unroll
  for (int f = 0; f < FIN_; ++f) acc += W1[j * FIN_ + f] * s_ins[f];
  s_x1[j] = fmaxf(acc, 0.0f);
  __syncthreads();
  float acc2 = b2[j];
#pragma unroll 8
  for (int k = 0; k < HID_; ++k) acc2 += W2[j * HID_ + k] * s_x1[k];
  ((unsigned short*)xbuf)[(size_t)bq * H_ + aq * HID_ + j] =
      f2bf(fmaxf(acc2, 0.0f));
}

// One wave computes the 4 gate tiles (i,f,g,o) of one 16x16 (M,N) tile and the
// full LSTM cell update for those elements.
__global__ void __launch_bounds__(32) lstm_layer(
    const bf16_t* __restrict__ x, const bf16_t* __restrict__ hin,
    bf16_t* __restrict__ hout, float* __restrict__ c,
    const bf16_t* __restrict__ WihF, const bf16_t* __restrict__ WhhF,
    const float* __restrict__ bih, const float* __restrict__ bhh) {
  int nt = blockIdx.x, mt = blockIdx.y;
  int lane = threadIdx.x, nl = lane & 15, hi = lane >> 4;
  int col = nt * 16 + nl;

  v8f acc[4];
#pragma unroll
  for (int g = 0; g < 4; ++g) {
    float b = bih[g * H_ + col] + bhh[g * H_ + col];
#pragma unroll
    for (int r = 0; r < 8; ++r) acc[g][r] = b;
  }

  int arow = mt * 16 + nl;
  for (int kt = 0; kt < KT; ++kt) {
    if (kt + 1 < KT) {  // global_prefetch_b8 of next activation chunk
      __builtin_prefetch(x   + (size_t)arow * H_ + (kt + 1) * 32, 0, 3);
      __builtin_prefetch(hin + (size_t)arow * H_ + (kt + 1) * 32, 0, 3);
    }
    v16bf ax = load_a(x,   arow, kt, hi);
    v16bf ah = load_a(hin, arow, kt, hi);
#pragma unroll
    for (int g = 0; g < 4; ++g) {
      v16bf bi = *(const v16bf*)(WihF +
          ((size_t)(g * NT_H + nt) * KT + kt) * 512 + lane * 16);
      acc[g] = __builtin_amdgcn_wmma_f32_16x16x32_bf16(
          false, ax, false, bi, (short)0, acc[g], false, false);
      v16bf bh = *(const v16bf*)(WhhF +
          ((size_t)(g * NT_H + nt) * KT + kt) * 512 + lane * 16);
      acc[g] = __builtin_amdgcn_wmma_f32_16x16x32_bf16(
          false, ah, false, bh, (short)0, acc[g], false, false);
    }
  }

  // Cell update: C/D layout gives element (m = r + hi*8, n = nl) in acc[*][r].
#pragma unroll
  for (int r = 0; r < 8; ++r) {
    int row = mt * 16 + r + hi * 8;
    size_t idx = (size_t)row * H_ + col;
    float ig = sigmoidf_(acc[0][r]);
    float fg = sigmoidf_(acc[1][r]);
    float gg = tanhf(acc[2][r]);
    float og = sigmoidf_(acc[3][r]);
    float cn = fg * c[idx] + ig * gg;
    c[idx] = cn;
    ((unsigned short*)hout)[idx] = f2bf(og * tanhf(cn));
  }
}

// y3 = relu(h1 @ W3.T + b3) -> bf16
__global__ void __launch_bounds__(32) head_gemm(
    const bf16_t* __restrict__ h1, const bf16_t* __restrict__ W3F,
    const float* __restrict__ b3, bf16_t* __restrict__ y3) {
  int nt = blockIdx.x, mt = blockIdx.y;
  int lane = threadIdx.x, nl = lane & 15, hi = lane >> 4;
  int col = nt * 16 + nl;
  v8f acc;
  float b = b3[col];
#pragma unroll
  for (int r = 0; r < 8; ++r) acc[r] = b;
  int arow = mt * 16 + nl;
  for (int kt = 0; kt < KT; ++kt) {
    if (kt + 1 < KT)
      __builtin_prefetch(h1 + (size_t)arow * H_ + (kt + 1) * 32, 0, 3);
    v16bf ax = load_a(h1, arow, kt, hi);
    v16bf bw = *(const v16bf*)(W3F + ((size_t)nt * KT + kt) * 512 + lane * 16);
    acc = __builtin_amdgcn_wmma_f32_16x16x32_bf16(
        false, ax, false, bw, (short)0, acc, false, false);
  }
#pragma unroll
  for (int r = 0; r < 8; ++r) {
    int row = mt * 16 + r + hi * 8;
    ((unsigned short*)y3)[(size_t)row * H_ + col] = f2bf(fmaxf(acc[r], 0.0f));
  }
}

// out = y3 @ W4.T + b4 + ins ; also feeds `prev` for post-burn-in steps.
__global__ void __launch_bounds__(128) head_out(
    const bf16_t* __restrict__ y3, const float* __restrict__ W4,
    const float* __restrict__ b4, const float* __restrict__ insbuf,
    float* __restrict__ out, float* __restrict__ prev, int t) {
  __shared__ float sy[H_];
  int b = blockIdx.x, tid = threadIdx.x;
  const unsigned short* yr = (const unsigned short*)y3 + (size_t)b * H_;
  for (int k = tid; k < H_; k += 128) sy[k] = bf2f(yr[k]);
  __syncthreads();
  if (tid < A_ * FIN_) {
    float acc = b4[tid];
    const float* wr = W4 + (size_t)tid * H_;
#pragma unroll 8
    for (int k = 0; k < H_; ++k) acc += wr[k] * sy[k];
    int a = tid >> 2, f = tid & 3;
    float v = acc + insbuf[((size_t)b * A_ + a) * FIN_ + f];
    out[(((size_t)b * A_ + a) * NSTEP + t) * FIN_ + f] = v;
    prev[((size_t)b * A_ + a) * FIN_ + f] = v;
  }
}

// ---------------- workspace layout (bytes) ----------------
constexpr size_t WF_LAYER_BYTES = (size_t)NT_G * KT * 512 * 2;  // 29,491,200
constexpr size_t W3F_BYTES      = (size_t)NT_H * KT * 512 * 2;  //  7,372,800
constexpr size_t ACT_BF16_BYTES = (size_t)B_ * H_ * 2;          //    491,520
constexpr size_t ACT_F32_BYTES  = (size_t)B_ * H_ * 4;          //    983,040
constexpr size_t INS_BYTES      = (size_t)B_ * A_ * FIN_ * 4;   //     61,440

constexpr size_t o_WihF = 0;
constexpr size_t o_WhhF = o_WihF + 2 * WF_LAYER_BYTES;
constexpr size_t o_W3F  = o_WhhF + 2 * WF_LAYER_BYTES;
constexpr size_t o_x    = o_W3F  + W3F_BYTES;
constexpr size_t o_h0   = o_x    + ACT_BF16_BYTES;       // 2 ping-pong bufs
constexpr size_t o_h1   = o_h0   + 2 * ACT_BF16_BYTES;   // 2 ping-pong bufs
constexpr size_t o_c0   = o_h1   + 2 * ACT_BF16_BYTES;
constexpr size_t o_c1   = o_c0   + ACT_F32_BYTES;
constexpr size_t o_y3   = o_c1   + ACT_F32_BYTES;
constexpr size_t o_ins  = o_y3   + ACT_BF16_BYTES;
constexpr size_t o_prev = o_ins  + INS_BYTES;
constexpr size_t WS_NEEDED = o_prev + INS_BYTES;         // ~130.4 MB

extern "C" void kernel_launch(void* const* d_in, const int* in_sizes, int n_in,
                              void* d_out, int out_size, void* d_ws,
                              size_t ws_size, hipStream_t stream) {
  (void)in_sizes; (void)n_in; (void)out_size;
  if (ws_size < WS_NEEDED) return;

  const float* inputs = (const float*)d_in[0];
  const float* W1  = (const float*)d_in[1];
  const float* b1  = (const float*)d_in[2];
  const float* W2  = (const float*)d_in[3];
  const float* b2  = (const float*)d_in[4];
  const float* Wih = (const float*)d_in[5];   // [L, 4H, H]
  const float* Whh = (const float*)d_in[6];   // [L, 4H, H]
  const float* bih = (const float*)d_in[7];   // [L, 4H]
  const float* bhh = (const float*)d_in[8];   // [L, 4H]
  const float* W3  = (const float*)d_in[9];   // [H, H]
  const float* b3  = (const float*)d_in[10];
  const float* W4  = (const float*)d_in[11];  // [A*FIN, H]
  const float* b4  = (const float*)d_in[12];
  const int* burn  = (const int*)d_in[14];    // burn_in_steps
  float* out = (float*)d_out;

  char* ws = (char*)d_ws;
  bf16_t* WihF = (bf16_t*)(ws + o_WihF);
  bf16_t* WhhF = (bf16_t*)(ws + o_WhhF);
  bf16_t* W3F  = (bf16_t*)(ws + o_W3F);
  bf16_t* xbuf = (bf16_t*)(ws + o_x);
  bf16_t* h0b  = (bf16_t*)(ws + o_h0);
  bf16_t* h1b  = (bf16_t*)(ws + o_h1);
  float*  c0   = (float*)(ws + o_c0);
  float*  c1   = (float*)(ws + o_c1);
  bf16_t* y3   = (bf16_t*)(ws + o_y3);
  float*  insb = (float*)(ws + o_ins);
  float*  prev = (float*)(ws + o_prev);

  // 1) Zero h/c/prev/activation region (fresh recurrent state each launch).
  size_t zero_u32 = (WS_NEEDED - o_x) / 4;
  zero_ws<<<2048, 256, 0, stream>>>((unsigned int*)(ws + o_x), zero_u32);

  // 2) One-time fp32 -> bf16 WMMA-fragment conversion (weights then live in L2).
  const size_t wlayer_f32 = (size_t)4 * H_ * H_;     // elems per layer
  const size_t wlayer_bf  = WF_LAYER_BYTES / 2;      // halves per layer
  int blk = 256;
  int grid_g = (NT_G * KT * 32 + blk - 1) / blk;
  int grid_h = (NT_H * KT * 32 + blk - 1) / blk;
  for (int l = 0; l < 2; ++l) {
    swizzle_w<<<grid_g, blk, 0, stream>>>(Wih + l * wlayer_f32,
                                          WihF + l * wlayer_bf, NT_G);
    swizzle_w<<<grid_g, blk, 0, stream>>>(Whh + l * wlayer_f32,
                                          WhhF + l * wlayer_bf, NT_G);
  }
  swizzle_w<<<grid_h, blk, 0, stream>>>(W3, W3F, NT_H);

  // 3) 64 sequential steps.
  const size_t act = (size_t)B_ * H_;  // elements per activation buffer
  for (int t = 0; t < NSTEP; ++t) {
    bf16_t* h0in  = h0b + (size_t)(t & 1) * act;
    bf16_t* h0out = h0b + (size_t)((t + 1) & 1) * act;
    bf16_t* h1in  = h1b + (size_t)(t & 1) * act;
    bf16_t* h1out = h1b + (size_t)((t + 1) & 1) * act;

    mlp_kernel<<<B_ * A_, HID_, 0, stream>>>(inputs, prev, W1, b1, W2, b2,
                                             xbuf, insb, burn, t);
    lstm_layer<<<dim3(NT_H, MT), 32, 0, stream>>>(
        xbuf, h0in, h0out, c0, WihF, WhhF, bih, bhh);
    lstm_layer<<<dim3(NT_H, MT), 32, 0, stream>>>(
        h0out, h1in, h1out, c1, WihF + wlayer_bf, WhhF + wlayer_bf,
        bih + 4 * H_, bhh + 4 * H_);
    head_gemm<<<dim3(NT_H, MT), 32, 0, stream>>>(h1out, W3F, b3, y3);
    head_out<<<B_, 128, 0, stream>>>(y3, W4, b4, insb, out, prev, t);
  }
}
